// DemoHybridAttention_86535001080333
// MI455X (gfx1250) — matrix operand order repaired and checked by reading, not compiled, not run on hardware
//
#include <hip/hip_runtime.h>

#define DIMD   512
#define NHEADS 16
#define HDIM   32
#define WIN2   128
#define MEMN   16
#define SEQ    2048
#define TTOT   (SEQ + MEMN)      /* 2064 */
#define NQT    (TTOT / 16)       /* 129 q/k tiles */
#define NEG_BIG (-1.0e30f)

typedef __attribute__((ext_vector_type(16))) __bf16 v16bf;
typedef __attribute__((ext_vector_type(8)))  float  v8f;

// Native conversion: selects gfx1250's v_cvt_pk_bf16_f32 when vectorized.
__device__ __forceinline__ unsigned short f32_bf16(float f) {
  __bf16 h = (__bf16)f;
  return __builtin_bit_cast(unsigned short, h);
}

union FragU { v16bf v; unsigned short s[16]; unsigned int u[8]; };

// A-matrix fragment (16x32 bf16) from a row-major bf16 tile at p=[row0][k0].
// lane L holds row (L&15); elem e -> k = (L>>4)*16 + e (contiguous, matches B frags).
// K-pairs are contiguous, so 8 dword loads (coalesce into b128 clauses).
__device__ __forceinline__ v16bf frag_a(const unsigned short* p, int ld, int lane) {
  FragU r;
  const unsigned short* rp = p + (lane & 15) * ld + ((lane >> 4) << 4);
#pragma unroll
  for (int i = 0; i < 4; ++i) r.u[i]     = *(const unsigned int*)(rp + 2 * i);
#pragma unroll
  for (int i = 0; i < 4; ++i) r.u[4 + i] = *(const unsigned int*)(rp + 8 + 2 * i);
  return r.v;
}

// B-matrix fragment (32x16 bf16) where B[k][n] = src[n][k] (transpose already
// in memory, "n-major"). lane L: col n = L&15, k = (L>>4)*16 + e -> contiguous.
__device__ __forceinline__ v16bf frag_b_t(const unsigned short* p, int ld, int lane) {
  FragU r;
  const unsigned short* rp = p + (lane & 15) * ld + ((lane >> 4) << 4);
#pragma unroll
  for (int i = 0; i < 8; ++i) r.u[i] = *(const unsigned int*)(rp + 2 * i);
  return r.v;
}

// B-matrix fragment from a k-major tile src[k][nbase+n] (strided over k, LDS).
__device__ __forceinline__ v16bf frag_b_k(const unsigned short* p, int ld, int lane, int nbase) {
  FragU r;
  const unsigned short* cp = p + nbase + (lane & 15);
  int kb = (lane >> 4) << 4;
#pragma unroll
  for (int e = 0; e < 16; ++e) r.s[e] = cp[(kb + e) * ld];
  return r.v;
}

__device__ __forceinline__ v8f wmma_bf16(v16bf a, v16bf b, v8f c) {
  return __builtin_amdgcn_wmma_f32_16x16x32_bf16(false, a, false, b, (short)0, c, false, false);
}

// ---------------------------------------------------------------------------
// K0a: convert + transpose the four 512x512 fp32 weights into bf16 (n-major).
// ---------------------------------------------------------------------------
__global__ __launch_bounds__(256) void wconv_kernel(const float* Wq, const float* Wk,
                                                    const float* Wv, const float* Wo,
                                                    unsigned short* Wt) {
  int m = blockIdx.y;
  const float* W = (m == 0) ? Wq : (m == 1) ? Wk : (m == 2) ? Wv : Wo;
  unsigned short* T = Wt + (size_t)m * DIMD * DIMD;
  int idx = blockIdx.x * blockDim.x + threadIdx.x;   // 0..262143
  int k = idx >> 9, n = idx & 511;                   // read coalesced over n
  T[(size_t)n * DIMD + k] = f32_bf16(W[(size_t)k * DIMD + n]);
}

// ---------------------------------------------------------------------------
// K0b: fused mem-concat + fp32->bf16 conversion of the activations, done ONCE
// so the projection GEMM streams pure bf16. Layout: Xin[z][t][d], z=which*2+b.
// Each thread converts a float2 pair -> one packed dword store.
// ---------------------------------------------------------------------------
__global__ __launch_bounds__(256) void xconv_kernel(const float* q, const float* k,
                                                    const float* v, const float* memp,
                                                    unsigned short* Xin) {
  int z = blockIdx.y;                              // which*2 + b
  int which = z >> 1, b = z & 1;
  const float* x = (which == 0) ? q : (which == 1) ? k : v;

  int pi = blockIdx.x * blockDim.x + threadIdx.x;  // pair index within one (z)
  int t = pi >> 8;                                 // 256 pairs per 512-wide row
  int d = (pi & 255) * 2;

  const float* src = (t < MEMN) ? (memp + (size_t)t * DIMD + d)
                                : (x + ((size_t)b * SEQ + (t - MEMN)) * DIMD + d);
  float2 f = *(const float2*)src;
  unsigned int pk = (unsigned int)f32_bf16(f.x) | ((unsigned int)f32_bf16(f.y) << 16);
  *(unsigned int*)(Xin + (size_t)z * TTOT * DIMD + (size_t)t * DIMD + d) = pk;
}

// ---------------------------------------------------------------------------
// K1: QKV projection GEMM, all-bf16 operands. One wave = one 16x16 output
// tile; block = 8 waves = 32 rows x 64 cols; K-loop 512 in steps of 32.
// ---------------------------------------------------------------------------
__global__ __launch_bounds__(256) void proj_kernel(
    const unsigned short* Xin, const unsigned short* Wt,
    const float* bq, const float* bk, const float* bv,
    unsigned short* Qh, unsigned short* Kh, unsigned short* Vh) {
  int z = blockIdx.z;
  int which = z >> 1, b = z & 1;
  const unsigned short* Xb   = Xin + (size_t)z * TTOT * DIMD;
  const unsigned short* W    = Wt + (size_t)which * DIMD * DIMD;
  const float*          bias = (which == 0) ? bq : (which == 1) ? bk : bv;
  unsigned short*       outp = (which == 0) ? Qh : (which == 1) ? Kh : Vh;

  int lane = threadIdx.x & 31, w = threadIdx.x >> 5;
  int ttile = blockIdx.y * 2 + (w >> 2);
  if (ttile >= NQT) return;                      // 129 row tiles, last block ragged
  int t0 = ttile * 16;
  int c0 = blockIdx.x * 64 + (w & 3) * 16;

  v8f acc = {};
#pragma unroll 4
  for (int k0 = 0; k0 < DIMD; k0 += 32) {
    __builtin_prefetch(Xb + (size_t)t0 * DIMD + k0 + 128, 0, 3);  // near-scope prefetch
    v16bf af = frag_a(Xb + (size_t)t0 * DIMD + k0, DIMD, lane);
    v16bf bf = frag_b_t(W + (size_t)c0 * DIMD + k0, DIMD, lane);
    acc = wmma_bf16(af, bf, acc);
  }
  int col = c0 + (lane & 15);
  float bval = bias[col];
  int h = col >> 5, dh = col & 31, hl = lane >> 4;
#pragma unroll
  for (int r = 0; r < 8; ++r) {
    int t = t0 + r + hl * 8;                     // C layout: M = r + 8*(lane>=16)
    outp[(((size_t)b * NHEADS + h) * TTOT + t) * HDIM + dh] = f32_bf16(acc[r] + bval);
  }
}

// ---------------------------------------------------------------------------
// K2: windowed attention with online (flash) softmax. One wave = one 16-row
// q-tile of one (b,h). K-tiles processed in pairs (32 keys) so P is a full
// 16x32 WMMA A-operand. V tiles DMA into LDS with the CDNA5 async engine
// (overlapped with the score WMMAs + softmax, gated by s_wait_asynccnt);
// P bounces C->A layout through per-wave LDS.
// ---------------------------------------------------------------------------
__global__ __launch_bounds__(256) void attn_kernel(
    const unsigned short* Qh, const unsigned short* Kh, const unsigned short* Vh,
    unsigned short* Ctx) {
  __shared__ unsigned short lds[8 * (32 * HDIM + 16 * 32)];   // per wave: V 32x32, P 16x32
  int lane = threadIdx.x & 31, w = threadIdx.x >> 5;
  int bh = blockIdx.y;
  int b = bh >> 4, h = bh & 15;
  int qt = blockIdx.x * 8 + w;
  if (qt >= NQT) return;                         // no barriers below -> safe exit
  int q0 = qt * 16;

  unsigned short* vbuf = lds + w * (32 * HDIM + 16 * 32);
  unsigned short* pbuf = vbuf + 32 * HDIM;

  const size_t headoff = ((size_t)(b * NHEADS + h)) * TTOT * HDIM;
  const unsigned short* Qb = Qh + headoff;
  const unsigned short* Kb = Kh + headoff;
  const unsigned short* Vb = Vh + headoff;

  v16bf aq = frag_a(Qb + (size_t)q0 * HDIM, HDIM, lane);   // head_dim==32: one A frag

  // k-tile schedule: mem tile 0 (global tokens) + window tiles [lo,hi]
  int lo, hi, extra;
  if (q0 < MEMN) { lo = 0; hi = NQT - 1; extra = 0; }      // mem queries see all
  else {
    lo = (q0 - WIN2) >> 4;       if (lo < 0) lo = 0;
    hi = (q0 + 15 + WIN2) >> 4;  if (hi > NQT - 1) hi = NQT - 1;
    extra = (lo > 0) ? 1 : 0;
  }
  int count = extra + (hi - lo + 1);

  float mrow[8], lrow[8];
#pragma unroll
  for (int r = 0; r < 8; ++r) { mrow[r] = NEG_BIG; lrow[r] = 0.0f; }
  v8f acc0 = {}, acc1 = {};
  int hl = lane >> 4, n = lane & 15;
  const float scale = 0.17677669529663687f;      // 1/sqrt(32)

  for (int i = 0; i < count; i += 2) {
    int kt0 = (extra && i == 0) ? 0 : (lo + i - extra);
    int kt1 = (i + 1 < count) ? (lo + (i + 1) - extra) : -1;

    // ---- stage 32 key rows of V into LDS (lane -> one 64B row) ----
#if __has_builtin(__builtin_amdgcn_global_load_async_to_lds_b128)
    {
      // Builtin params are int4* (vector_size 16) in AS1 (global) / AS3 (LDS);
      // construct via integer->pointer casts (AS3 value = LDS byte offset =
      // low 32 bits of the generic LDS address per the aperture mapping).
      typedef int v4i_vs __attribute__((vector_size(16)));
      typedef __attribute__((address_space(1))) v4i_vs* gv4p;
      typedef __attribute__((address_space(3))) v4i_vs* lv4p;
      int rr = lane;
      // odd tail: clamp to a valid tile; its P columns are exactly 0 so the
      // (finite) values never contribute to the P*V accumulation.
      int kt = (rr < 16) ? kt0 : ((kt1 >= 0) ? kt1 : kt0);
      const unsigned short* src = Vb + (size_t)(kt * 16 + (rr & 15)) * HDIM;
      gv4p ga = (gv4p)(unsigned long long)(const void*)src;
      lv4p la = (lv4p)(unsigned int)(unsigned long long)(void*)(vbuf + rr * HDIM);
      // INST_OFFSET is added to both global and LDS addresses (ISA 08 §4.4)
      __builtin_amdgcn_global_load_async_to_lds_b128(ga, la, 0, 0);
      __builtin_amdgcn_global_load_async_to_lds_b128(ga, la, 16, 0);
      __builtin_amdgcn_global_load_async_to_lds_b128(ga, la, 32, 0);
      __builtin_amdgcn_global_load_async_to_lds_b128(ga, la, 48, 0);
    }
#else
    {
      int rr = lane;
      int kt = (rr < 16) ? kt0 : kt1;
      unsigned int* dst = (unsigned int*)(vbuf + rr * HDIM);
      if (kt >= 0) {
        const unsigned int* src = (const unsigned int*)(Vb + (size_t)(kt * 16 + (rr & 15)) * HDIM);
#pragma unroll
        for (int c = 0; c < HDIM / 2; ++c) dst[c] = src[c];
      } else {
#pragma unroll
        for (int c = 0; c < HDIM / 2; ++c) dst[c] = 0u;
      }
    }
#endif

    // scores: Q (16x32) x K^T (32x16) -- K rows load with the n-major pattern
    v8f c0 = {}, c1 = {};
    c0 = wmma_bf16(aq, frag_b_t(Kb + (size_t)(kt0 * 16) * HDIM, HDIM, lane), c0);
    if (kt1 >= 0)
      c1 = wmma_bf16(aq, frag_b_t(Kb + (size_t)(kt1 * 16) * HDIM, HDIM, lane), c1);

    // online softmax update, per C-layout row (r + 8*hl), col n across 16 lanes
#pragma unroll
    for (int r = 0; r < 8; ++r) {
      int qi = q0 + r + hl * 8;
      int kj0 = kt0 * 16 + n;
      int d0 = qi - kj0;
      float s0 = c0[r] * scale;
      if (!((qi < MEMN) | (kj0 < MEMN) | ((d0 <= WIN2) & (d0 >= -WIN2)))) s0 = NEG_BIG;
      float s1 = NEG_BIG;
      if (kt1 >= 0) {
        int kj1 = kt1 * 16 + n;
        int d1 = qi - kj1;
        s1 = c1[r] * scale;
        if (!((qi < MEMN) | (kj1 < MEMN) | ((d1 <= WIN2) & (d1 >= -WIN2)))) s1 = NEG_BIG;
      }
      float mx = fmaxf(s0, s1);
#pragma unroll
      for (int d = 1; d < 16; d <<= 1) mx = fmaxf(mx, __shfl_xor(mx, d));  // stays in half-wave
      float mn = fmaxf(mrow[r], mx);
      float alpha = __expf(mrow[r] - mn);
      float p0 = __expf(s0 - mn);
      float p1 = __expf(s1 - mn);
      float rs = p0 + p1;
#pragma unroll
      for (int d = 1; d < 16; d <<= 1) rs += __shfl_xor(rs, d);
      lrow[r] = lrow[r] * alpha + rs;
      mrow[r] = mn;
      acc0[r] *= alpha;
      acc1[r] *= alpha;
      pbuf[(r + hl * 8) * 32 + n]      = f32_bf16(p0);     // C-layout -> LDS
      pbuf[(r + hl * 8) * 32 + 16 + n] = f32_bf16(p1);
    }

    // V tile must be resident in LDS before the k-major B gather
#if __has_builtin(__builtin_amdgcn_global_load_async_to_lds_b128)
#if __has_builtin(__builtin_amdgcn_s_wait_asynccnt)
    __builtin_amdgcn_s_wait_asynccnt(0);
#else
    asm volatile("s_wait_asynccnt 0x0" ::: "memory");
#endif
#endif

    // ctx += P (16x32) x V (32x32, two 16-col halves)
    v16bf ap  = frag_a(pbuf, 32, lane);                    // LDS -> A layout
    acc0 = wmma_bf16(ap, frag_b_k(vbuf, HDIM, lane, 0),  acc0);
    acc1 = wmma_bf16(ap, frag_b_k(vbuf, HDIM, lane, 16), acc1);
  }

  // normalize and store ctx, dropping the MEM prefix rows
#pragma unroll
  for (int r = 0; r < 8; ++r) {
    int qi = q0 + r + hl * 8;
    if (qi >= MEMN) {
      float inv = 1.0f / lrow[r];
      size_t base = ((size_t)b * SEQ + (qi - MEMN)) * DIMD + h * HDIM;
      Ctx[base + n]      = f32_bf16(acc0[r] * inv);
      Ctx[base + 16 + n] = f32_bf16(acc1[r] * inv);
    }
  }
}

// ---------------------------------------------------------------------------
// K3: output projection ctx(B*S x 512) @ Wo + bo -> fp32 d_out.
// ---------------------------------------------------------------------------
__global__ __launch_bounds__(256) void outproj_kernel(
    const unsigned short* Ctx, const unsigned short* Wto, const float* bo, float* out) {
  int lane = threadIdx.x & 31, w = threadIdx.x >> 5;
  int r0 = (blockIdx.y * 2 + (w >> 2)) * 16;     // flat row in [0, 4096)
  int c0 = blockIdx.x * 64 + (w & 3) * 16;
  v8f acc = {};
#pragma unroll 4
  for (int k0 = 0; k0 < DIMD; k0 += 32) {
    v16bf af = frag_a(Ctx + (size_t)r0 * DIMD + k0, DIMD, lane);
    v16bf bf = frag_b_t(Wto + (size_t)c0 * DIMD + k0, DIMD, lane);
    acc = wmma_bf16(af, bf, acc);
  }
  int col = c0 + (lane & 15);
  float bval = bo[col];
  int hl = lane >> 4;
#pragma unroll
  for (int r = 0; r < 8; ++r) {
    int row = r0 + r + hl * 8;
    out[(size_t)row * DIMD + col] = acc[r] + bval;
  }
}

// ---------------------------------------------------------------------------
extern "C" void kernel_launch(void* const* d_in, const int* in_sizes, int n_in,
                              void* d_out, int out_size, void* d_ws, size_t ws_size,
                              hipStream_t stream) {
  (void)in_sizes; (void)n_in; (void)out_size; (void)ws_size;
  const float* query = (const float*)d_in[0];
  const float* key   = (const float*)d_in[1];
  const float* value = (const float*)d_in[2];
  const float* memp  = (const float*)d_in[3];
  const float* Wq    = (const float*)d_in[4];
  const float* bq    = (const float*)d_in[5];
  const float* Wk    = (const float*)d_in[6];
  const float* bk    = (const float*)d_in[7];
  const float* Wv    = (const float*)d_in[8];
  const float* bv    = (const float*)d_in[9];
  const float* Wo    = (const float*)d_in[10];
  const float* bo    = (const float*)d_in[11];
  float* out = (float*)d_out;

  // workspace carve-up (bf16 stored as ushort); ~32 MB total, fits in L2
  unsigned short* wsp = (unsigned short*)d_ws;
  unsigned short* Wt  = wsp;                               // 4 * 512*512
  unsigned short* Xin = Wt + 4 * DIMD * DIMD;              // 6 * 2064*512
  unsigned short* Qh  = Xin + 6 * (size_t)TTOT * DIMD;     // 2*16*2064*32
  unsigned short* Kh  = Qh + 2 * NHEADS * TTOT * HDIM;
  unsigned short* Vh  = Kh + 2 * NHEADS * TTOT * HDIM;
  unsigned short* Ctx = Vh + 2 * NHEADS * TTOT * HDIM;     // 2*2048*512

  wconv_kernel<<<dim3(1024, 4), 256, 0, stream>>>(Wq, Wk, Wv, Wo, Wt);
  xconv_kernel<<<dim3((TTOT * DIMD / 2) / 256, 6), 256, 0, stream>>>(
      query, key, value, memp, Xin);
  proj_kernel<<<dim3(DIMD / 64, (NQT + 1) / 2, 6), 256, 0, stream>>>(
      Xin, Wt, bq, bk, bv, Qh, Kh, Vh);
  attn_kernel<<<dim3((NQT + 7) / 8, 2 * NHEADS), 256, 0, stream>>>(Qh, Kh, Vh, Ctx);
  outproj_kernel<<<dim3(DIMD / 64, (2 * SEQ) / 32), 256, 0, stream>>>(
      Ctx, Wt + 3 * DIMD * DIMD, bo, out);
}
